// SlotConditioner_12859132084466
// MI455X (gfx1250) — matrix operand order repaired and checked by reading, not compiled, not run on hardware
//
#include <hip/hip_runtime.h>
#include <hip/hip_bf16.h>

typedef _Float16 v16h __attribute__((ext_vector_type(16)));
typedef _Float16 v8h  __attribute__((ext_vector_type(8)));
typedef float    v8f  __attribute__((ext_vector_type(8)));

#define BATCH 1024
#define NS    128
#define SD    64
#define TD    512
#define AD    384
#define HID   256
#define OD    64
#define M_LO  40
#define NM    16

// Exact-form gelu with branch-free erf (Abramowitz-Stegun 7.1.26, |err|<=1.5e-7,
// far below the f16 quantization applied right after).  Uses hw v_exp_f32 and
// v_rcp_f32 instead of ocml's branchy erff -> ~2x fewer VALU ops on the
// dominant cost of this workload.
__device__ __forceinline__ float gelu_exact(float x) {
    const float ax = __builtin_fabsf(x);
    const float t  = __builtin_amdgcn_rcpf(__builtin_fmaf(0.3275911f, ax, 1.0f));
    float p = __builtin_fmaf(1.061405429f, t, -1.453152027f);
    p = __builtin_fmaf(p, t, 1.421413741f);
    p = __builtin_fmaf(p, t, -0.284496736f);
    p = __builtin_fmaf(p, t, 0.254829592f);
    p = p * t;
    const float e   = __expf(-ax * ax);          // v_exp_f32
    const float era = __builtin_fmaf(-p, e, 1.0f);
    const float erf = __builtin_copysignf(era, x);
    return 0.5f * x * (1.0f + erf);
}

// out[n*K + k] = (f16) in[k*N + n]   (transpose + downconvert weights once)
__global__ void transpose_f32_to_f16(const float* __restrict__ in,
                                     _Float16* __restrict__ out, int K, int N) {
    int idx = blockIdx.x * blockDim.x + threadIdx.x;
    if (idx >= K * N) return;
    int k = idx / N, n = idx - k * N;
    out[(size_t)n * K + k] = (_Float16)in[(size_t)k * N + n];
}

// C[M x 256] = A[M x K](f32, row-major, lda=K) @ B, where B is given transposed
// as Bt[N x ldb] f16 with column offset boff (so B[k][n] = Bt[n*ldb + boff + k]).
// One wave computes a 16(M) x 64(N) tile; grid.y picks the 64-wide N group.
__global__ void gemm_wmma_f16(const float* __restrict__ A, int K,
                              const _Float16* __restrict__ Bt, int ldb, int boff,
                              float* __restrict__ C, int M,
                              const float* __restrict__ bias, int add_bias) {
    const int wave  = threadIdx.x >> 5;
    const int l     = threadIdx.x & 31;
    const int m     = l & 15;      // row within tile (A) / col within tile (B,C)
    const int kh    = l >> 4;      // which K-half group this lane feeds
    const int mtile = blockIdx.x * (blockDim.x >> 5) + wave;
    if (mtile * 16 >= M) return;
    const int n0 = blockIdx.y * 64;

    const float* Arow = A + (size_t)(mtile * 16 + m) * K;
    v8f acc[4] = {};

    for (int kb = 0; kb < K; kb += 32) {
        // A fragment per ISA 16-bit A 16x32 layout:
        // lane kh=0: K = kb+0..7 (v0-3), kb+16..23 (v4-7); kh=1: +8 on both.
        const int k0 = kb + kh * 8;
        v16h a;
        #pragma unroll
        for (int i = 0; i < 8; ++i) {
            a[i]     = (_Float16)Arow[k0 + i];
            a[i + 8] = (_Float16)Arow[k0 + 16 + i];
        }
        #pragma unroll
        for (int t = 0; t < 4; ++t) {
            // B fragment: lane col n = m, K = kb+kh*8+0..7 and kb+16+kh*8+0..7
            const v8h* bv = (const v8h*)(Bt + (size_t)(n0 + t * 16 + m) * ldb
                                            + boff + kb + kh * 8);
            v8h lo = bv[0], hi = bv[2];
            v16h bf;
            #pragma unroll
            for (int i = 0; i < 8; ++i) { bf[i] = lo[i]; bf[i + 8] = hi[i]; }
            acc[t] = __builtin_amdgcn_wmma_f32_16x16x32_f16(
                false, a, false, bf, (short)0, acc[t], false, false);
        }
    }
    // C/D layout: lane col = m, VGPR r holds row r + 8*kh
    #pragma unroll
    for (int t = 0; t < 4; ++t) {
        #pragma unroll
        for (int r = 0; r < 8; ++r) {
            const int mr = mtile * 16 + r + 8 * kh;
            const int nc = n0 + t * 16 + m;
            float v = acc[t][r];
            if (add_bias) v += bias[nc];
            C[(size_t)mr * HID + nc] = v;
        }
    }
}

// Stage the full 64x256 f16 W2t (32 KB) into LDS once per block (8 waves reuse
// it 8x each); inner-loop B fragments then come from ds_load_b128 instead of
// redundant global loads, keeping the vmem pipe free for activation reads.
__device__ __forceinline__ void stage_w2(const _Float16* __restrict__ W2t,
                                         _Float16* sW) {
    const float4* src = (const float4*)W2t;
    float4*       dst = (float4*)sW;
    // 64*256 halves = 2048 float4 chunks; 256 threads x 8
    #pragma unroll
    for (int i = 0; i < 8; ++i)
        dst[threadIdx.x + i * 256] = src[threadIdx.x + i * 256];
    __syncthreads();
}

__device__ __forceinline__ v16h lds_bfrag(const _Float16* sW, int row, int col) {
    const v8h* bv = (const v8h*)(sW + row * HID + col);
    v8h lo = bv[0], hi = bv[2];
    v16h bf;
    #pragma unroll
    for (int i = 0; i < 8; ++i) { bf[i] = lo[i]; bf[i + 8] = hi[i]; }
    return bf;
}

// Fused layer-2 (main path): h = gelu(S[slot] + T[b]) built per-lane directly
// in WMMA A-fragment layout, then out = h @ W2t + b2.  Tile = 16 slots of one b.
__global__ void fused_main(const float* __restrict__ S,      // [128][256] (b1m folded)
                           const float* __restrict__ T,      // [1024][256]
                           const _Float16* __restrict__ W2t, // [64][256] f16 transposed
                           const float* __restrict__ b2,
                           float* __restrict__ out) {        // [1024][128][64]
    __shared__ __align__(16) _Float16 sW[OD * HID];
    stage_w2(W2t, sW);

    const int wave = threadIdx.x >> 5;
    const int l    = threadIdx.x & 31;
    const int m    = l & 15;
    const int kh   = l >> 4;
    const int tile = blockIdx.x * (blockDim.x >> 5) + wave;  // 0..8191
    const int b     = tile >> 3;
    const int slot0 = (tile & 7) << 4;

    const float* Srow = S + (size_t)(slot0 + m) * HID;
    const float* Trow = T + (size_t)b * HID;
    v8f acc[4] = {};

    #pragma unroll
    for (int kb = 0; kb < HID; kb += 32) {
        const int k0 = kb + kh * 8;
        v16h a;
        #pragma unroll
        for (int i = 0; i < 8; ++i) {
            a[i]     = (_Float16)gelu_exact(Srow[k0 + i]      + Trow[k0 + i]);
            a[i + 8] = (_Float16)gelu_exact(Srow[k0 + 16 + i] + Trow[k0 + 16 + i]);
        }
        #pragma unroll
        for (int t = 0; t < 4; ++t) {
            v16h bf = lds_bfrag(sW, t * 16 + m, kb + kh * 8);
            acc[t] = __builtin_amdgcn_wmma_f32_16x16x32_f16(
                false, a, false, bf, (short)0, acc[t], false, false);
        }
    }
    float* orow = out + ((size_t)b * NS + slot0) * OD;
    #pragma unroll
    for (int t = 0; t < 4; ++t) {
        const float bb = b2[t * 16 + m];
        #pragma unroll
        for (int r = 0; r < 8; ++r)
            orow[(size_t)(r + 8 * kh) * OD + t * 16 + m] = acc[t][r] + bb;
    }
}

// Fused layer-2 (mouth path): tile = all 16 mouth slots of one b.
__global__ void fused_mouth(const float* __restrict__ Su,     // [16][256] (b1u folded)
                            const float* __restrict__ Tu,     // [1024][256]
                            const float* __restrict__ Au,     // [1024][256]
                            const _Float16* __restrict__ W2t, // [64][256]
                            const float* __restrict__ b2,
                            float* __restrict__ out) {        // [1024][128][64]
    __shared__ __align__(16) _Float16 sW[OD * HID];
    stage_w2(W2t, sW);

    const int wave = threadIdx.x >> 5;
    const int l    = threadIdx.x & 31;
    const int m    = l & 15;
    const int kh   = l >> 4;
    const int b    = blockIdx.x * (blockDim.x >> 5) + wave;   // 0..1023

    const float* Srow = Su + (size_t)m * HID;
    const float* Trow = Tu + (size_t)b * HID;
    const float* Arow = Au + (size_t)b * HID;
    v8f acc[4] = {};

    #pragma unroll
    for (int kb = 0; kb < HID; kb += 32) {
        const int k0 = kb + kh * 8;
        v16h a;
        #pragma unroll
        for (int i = 0; i < 8; ++i) {
            a[i]     = (_Float16)gelu_exact(Srow[k0 + i]      + Trow[k0 + i]      + Arow[k0 + i]);
            a[i + 8] = (_Float16)gelu_exact(Srow[k0 + 16 + i] + Trow[k0 + 16 + i] + Arow[k0 + 16 + i]);
        }
        #pragma unroll
        for (int t = 0; t < 4; ++t) {
            v16h bf = lds_bfrag(sW, t * 16 + m, kb + kh * 8);
            acc[t] = __builtin_amdgcn_wmma_f32_16x16x32_f16(
                false, a, false, bf, (short)0, acc[t], false, false);
        }
    }
    float* orow = out + ((size_t)b * NS + M_LO) * OD;
    #pragma unroll
    for (int t = 0; t < 4; ++t) {
        const float bb = b2[t * 16 + m];
        #pragma unroll
        for (int r = 0; r < 8; ++r)
            orow[(size_t)(r + 8 * kh) * OD + t * 16 + m] = acc[t][r] + bb;
    }
}

extern "C" void kernel_launch(void* const* d_in, const int* in_sizes, int n_in,
                              void* d_out, int out_size, void* d_ws, size_t ws_size,
                              hipStream_t stream) {
    (void)in_sizes; (void)n_in; (void)out_size; (void)ws_size;
    const float* text  = (const float*)d_in[0];
    const float* audio = (const float*)d_in[1];
    const float* emb   = (const float*)d_in[2];
    const float* w1m   = (const float*)d_in[3];
    const float* b1m   = (const float*)d_in[4];
    const float* w2m   = (const float*)d_in[5];
    const float* b2m   = (const float*)d_in[6];
    const float* w1u   = (const float*)d_in[7];
    const float* b1u   = (const float*)d_in[8];
    const float* w2u   = (const float*)d_in[9];
    const float* b2u   = (const float*)d_in[10];
    float* out = (float*)d_out;

    char* ws = (char*)d_ws;
    size_t off = 0;
    auto carve = [&](size_t bytes) -> void* {
        void* p = ws + off;
        off = (off + bytes + 255) & ~(size_t)255;
        return p;
    };
    _Float16* w1mt = (_Float16*)carve((size_t)(SD + TD) * HID * 2);      // [256][576]
    _Float16* w1ut = (_Float16*)carve((size_t)(SD + TD + AD) * HID * 2); // [256][960]
    _Float16* w2mt = (_Float16*)carve((size_t)HID * OD * 2);             // [64][256]
    _Float16* w2ut = (_Float16*)carve((size_t)HID * OD * 2);             // [64][256]
    float* Sb  = (float*)carve((size_t)NS * HID * 4);                    // [128][256]
    float* Sub = (float*)carve((size_t)NM * HID * 4);                    // [16][256]
    float* Tb  = (float*)carve((size_t)BATCH * HID * 4);                 // [1024][256]
    float* Tub = (float*)carve((size_t)BATCH * HID * 4);
    float* Aub = (float*)carve((size_t)BATCH * HID * 4);

    // 1) Transpose + f16-convert all weight matrices (tiny, one-shot each call).
    transpose_f32_to_f16<<<((SD+TD)*HID + 255) / 256, 256, 0, stream>>>(w1m, w1mt, SD+TD, HID);
    transpose_f32_to_f16<<<((SD+TD+AD)*HID + 255) / 256, 256, 0, stream>>>(w1u, w1ut, SD+TD+AD, HID);
    transpose_f32_to_f16<<<(HID*OD + 255) / 256, 256, 0, stream>>>(w2m, w2mt, HID, OD);
    transpose_f32_to_f16<<<(HID*OD + 255) / 256, 256, 0, stream>>>(w2u, w2ut, HID, OD);

    // 2) Factored layer-1 GEMMs (broadcast structure exploited).
    //    T  = text  @ w1m[64:576]          (1024x512x256)
    //    Tu = text  @ w1u[64:576]          (1024x512x256)
    //    Au = audio @ w1u[576:960]         (1024x384x256)
    //    S  = emb   @ w1m[0:64]  + b1m     (128x64x256)
    //    Su = emb[40:56] @ w1u[0:64] + b1u (16x64x256)
    gemm_wmma_f16<<<dim3(16, 4), 128, 0, stream>>>(text,  TD, w1mt, SD+TD,    SD, Tb,  BATCH, nullptr, 0);
    gemm_wmma_f16<<<dim3(16, 4), 128, 0, stream>>>(text,  TD, w1ut, SD+TD+AD, SD, Tub, BATCH, nullptr, 0);
    gemm_wmma_f16<<<dim3(16, 4), 128, 0, stream>>>(audio, AD, w1ut, SD+TD+AD, SD+TD, Aub, BATCH, nullptr, 0);
    gemm_wmma_f16<<<dim3(2, 4), 128, 0, stream>>>(emb, SD, w1mt, SD+TD, 0, Sb, NS, b1m, 1);
    gemm_wmma_f16<<<dim3(1, 4), 128, 0, stream>>>(emb + M_LO * SD, SD, w1ut, SD+TD+AD, 0, Sub, NM, b1u, 1);

    // 3) Fused gelu + layer-2 (bulk of the FLOPs), then mouth overwrite (ordered
    //    after fused_main by stream order).
    fused_main<<<BATCH * (NS / 16) / 8, 256, 0, stream>>>(Sb, Tb, w2mt, b2m, out);
    fused_mouth<<<BATCH / 8, 256, 0, stream>>>(Sub, Tub, Aub, w2ut, b2u, out);
}
